// Decoder_3539053052711
// MI455X (gfx1250) — matrix operand order repaired and checked by reading, not compiled, not run on hardware
//
#include <hip/hip_runtime.h>
#include <math.h>

// ---------------------------------------------------------------------------
// Problem dims
// ---------------------------------------------------------------------------
#define V   32000
#define D   1024
#define H   1024
#define E   1024
#define S   4096
#define NT  64      // number of 16-wide N tiles over H in the attention GEMM

// d_out layout (floats): output[V] | next_h[H] | next_c[H] | attn[S]
#define OUT_LOGP 0
#define OUT_H    (V)
#define OUT_C    (V + H)
#define OUT_ATTN (V + 2 * H)

// workspace layout (float offsets)
#define WS_GATES   0          // 4096
#define WS_HT      4096       // 1024
#define WS_HPART   5120       // 1024
#define WS_SCORES  6144       // 4096
#define WS_ATTN    10240      // 4096
#define WS_CT      14336      // 1024
#define WS_HTILDA  15360      // 1024
#define WS_CTPART  16384      // 64*1024
#define WS_PARTIAL 81920      // 64*4096
#define WS_LOGITS  344064     // 32000
// bf16 region (byte offset, 32B aligned, past 376064 floats = 1504256 B)
#define WS_BF_BYTES (1536u * 1024u)
#define WS_ENCB_OFF 0                      // 4096*1024 ushorts
#define WS_WSEB_OFF (4096u * 1024u)        // 1024*1024 ushorts

typedef __attribute__((ext_vector_type(16))) __bf16          v16bf;
typedef __attribute__((ext_vector_type(8)))  float           v8f;
typedef __attribute__((ext_vector_type(8)))  unsigned short  ush8;
typedef __attribute__((ext_vector_type(16))) unsigned short  ush16;

static __device__ __forceinline__ float wave_sum32(float v) {
    v += __shfl_xor(v, 16, 32);
    v += __shfl_xor(v, 8, 32);
    v += __shfl_xor(v, 4, 32);
    v += __shfl_xor(v, 2, 32);
    v += __shfl_xor(v, 1, 32);
    return v;
}
static __device__ __forceinline__ float half_sum16(float v) {
    v += __shfl_xor(v, 8, 32);
    v += __shfl_xor(v, 4, 32);
    v += __shfl_xor(v, 2, 32);
    v += __shfl_xor(v, 1, 32);
    return v;
}
static __device__ __forceinline__ float sigf(float x) { return 1.0f / (1.0f + __expf(-x)); }
static __device__ __forceinline__ unsigned short f2bf(float f) {
    unsigned int u = __float_as_uint(f);
    unsigned int r = u + 0x7FFFu + ((u >> 16) & 1u);   // round-to-nearest-even
    return (unsigned short)(r >> 16);
}
static __device__ __forceinline__ float dot4(float4 a, float4 b) {
    return a.x * b.x + a.y * b.y + a.z * b.z + a.w * b.w;
}

// ---------------------------------------------------------------------------
// K1: gates = W_ih @ x + b_ih + W_hh @ h_prev + b_hh    (one wave per row)
// ---------------------------------------------------------------------------
__global__ void k_gates(const int* __restrict__ symbol,
                        const float* __restrict__ emb,
                        const float* __restrict__ Wih, const float* __restrict__ bih,
                        const float* __restrict__ Whh, const float* __restrict__ bhh,
                        const float* __restrict__ h0,
                        float* __restrict__ gates) {
    const int lane = threadIdx.x & 31;
    const int row  = (blockIdx.x * blockDim.x + threadIdx.x) >> 5;   // 0..1023
    const float4* x4  = (const float4*)(emb + (size_t)symbol[0] * D);
    const float4* h4  = (const float4*)h0;
    const float4* wi4 = (const float4*)(Wih + (size_t)row * D);
    const float4* wh4 = (const float4*)(Whh + (size_t)row * H);
    float acc = 0.0f;
#pragma unroll
    for (int j = 0; j < 8; ++j) {
        int idx = j * 32 + lane;
        acc += dot4(wi4[idx], x4[idx]);
        acc += dot4(wh4[idx], h4[idx]);
    }
    acc = wave_sum32(acc);
    if (lane == 0) gates[row] = acc + bih[row] + bhh[row];
}

// ---------------------------------------------------------------------------
// K2: LSTM pointwise -> next_h, next_c, ws_ht
// ---------------------------------------------------------------------------
__global__ void k_lstm_pw(const float* __restrict__ gates,
                          const float* __restrict__ c0,
                          float* __restrict__ out_h, float* __restrict__ out_c,
                          float* __restrict__ ws_ht) {
    int k = blockIdx.x * blockDim.x + threadIdx.x;
    if (k >= H) return;
    float i = gates[k], f = gates[H + k], g = gates[2 * H + k], o = gates[3 * H + k];
    float c = sigf(f) * c0[k] + sigf(i) * tanhf(g);
    float h = sigf(o) * tanhf(c);
    out_c[k] = c;
    out_h[k] = h;
    ws_ht[k] = h;
}

// ---------------------------------------------------------------------------
// K3: h_part[r] = W_score[r, 0:H] . ht + b_score[r]   (one wave per row)
// ---------------------------------------------------------------------------
__global__ void k_hpart(const float* __restrict__ Wscore, const float* __restrict__ bscore,
                        const float* __restrict__ ht, float* __restrict__ hpart) {
    const int lane = threadIdx.x & 31;
    const int row  = (blockIdx.x * blockDim.x + threadIdx.x) >> 5;   // 0..255
    const float4* w4 = (const float4*)(Wscore + (size_t)row * (H + E));
    const float4* h4 = (const float4*)ht;
    float acc = 0.0f;
#pragma unroll
    for (int j = 0; j < 8; ++j) {
        int idx = j * 32 + lane;
        acc += dot4(w4[idx], h4[idx]);
    }
    acc = wave_sum32(acc);
    if (lane == 0) hpart[row] = acc + bscore[row];
}

// ---------------------------------------------------------------------------
// K4: fp32 -> bf16 conversion of encoder_states and W_score[:, H:]
// ---------------------------------------------------------------------------
__global__ void k_convert(const float* __restrict__ enc, const float* __restrict__ Wscore,
                          unsigned short* __restrict__ encb, unsigned short* __restrict__ wseb) {
    const unsigned total = S * E + H * E;
    for (unsigned i = blockIdx.x * blockDim.x + threadIdx.x; i < total;
         i += gridDim.x * blockDim.x) {
        if (i < S * E) {
            encb[i] = f2bf(enc[i]);
        } else {
            unsigned j = i - S * E;           // j = r*E + k
            unsigned r = j >> 10, k = j & 1023;
            wseb[j] = f2bf(Wscore[(size_t)r * (H + E) + H + k]);
        }
    }
}

// ---------------------------------------------------------------------------
// K5: attention GEMM (WMMA bf16, f32 acc) with fused tanh + w_va epilogue.
//   acc[s,n] = sum_k enc[s,k] * Wse[n,k]
//   partial[nTile, s] = sum_{n in tile} tanh(acc + hpart[n]) * w_va[n]
// Register-blocked: each wave owns a 32x64 macro-tile (2 M-tiles x 4 N-tiles,
// 8 accumulators). Per K step: 4 A loads + 8 B loads feed 8 WMMAs
// (1.5 b128 loads per WMMA), cutting L2 fragment traffic ~2.6x vs 1 tile/wave.
// ---------------------------------------------------------------------------
__global__ void k_attn_gemm(const unsigned short* __restrict__ encb,
                            const unsigned short* __restrict__ wseb,
                            const float* __restrict__ hpart,
                            const float* __restrict__ wva,
                            float* __restrict__ partial) {
    const int lane   = threadIdx.x & 31;
    const int wave   = (blockIdx.x * blockDim.x + threadIdx.x) >> 5;
    const int sGroup = wave >> 4;         // 0..127  (rows 32*sGroup .. +31)
    const int nGroup = wave & 15;         // 0..15   (cols 64*nGroup .. +63)
    const int ln     = lane & 15;         // fragment row (A: M, B: N)
    const int kg     = lane >> 4;         // K-group select

    // A rows for the two M-tiles of this macro-tile
    const unsigned short* aRow0 = encb + (size_t)(sGroup * 32 + ln) * E;
    const unsigned short* aRow1 = aRow0 + (size_t)16 * E;
    // B rows for the four N-tiles
    const unsigned short* bRow0 = wseb + (size_t)(nGroup * 64 + ln) * E;
    const unsigned short* bRow1 = bRow0 + (size_t)16 * E;
    const unsigned short* bRow2 = bRow0 + (size_t)32 * E;
    const unsigned short* bRow3 = bRow0 + (size_t)48 * E;

    // per-lane column bias / score weight for each N-tile (n = lane&15)
    float hp[4], wv[4];
#pragma unroll
    for (int j = 0; j < 4; ++j) {
        hp[j] = hpart[nGroup * 64 + j * 16 + ln];
        wv[j] = wva[nGroup * 64 + j * 16 + ln];
    }

    v8f acc[2][4] = {};
#pragma unroll 2
    for (int k0 = 0; k0 < E; k0 += 32) {
        // A fragments: lane(m, kg) holds K = kg*16..+15 (contiguous)
        ush8 a00 = *(const ush8*)(aRow0 + k0 + kg * 16);
        ush8 a01 = *(const ush8*)(aRow0 + k0 + 8 + kg * 16);
        ush8 a10 = *(const ush8*)(aRow1 + k0 + kg * 16);
        ush8 a11 = *(const ush8*)(aRow1 + k0 + 8 + kg * 16);
        ush16 aU[2];
        aU[0] = __builtin_shufflevector(a00, a01, 0, 1, 2, 3, 4, 5, 6, 7,
                                                  8, 9, 10, 11, 12, 13, 14, 15);
        aU[1] = __builtin_shufflevector(a10, a11, 0, 1, 2, 3, 4, 5, 6, 7,
                                                  8, 9, 10, 11, 12, 13, 14, 15);
        // B fragments: lane(n, kg) holds K = kg*16..+15 (contiguous, 32B aligned)
        ush16 bU[4];
        bU[0] = *(const ush16*)(bRow0 + k0 + kg * 16);
        bU[1] = *(const ush16*)(bRow1 + k0 + kg * 16);
        bU[2] = *(const ush16*)(bRow2 + k0 + kg * 16);
        bU[3] = *(const ush16*)(bRow3 + k0 + kg * 16);

        if (k0 + 128 < E) {
            __builtin_prefetch(aRow0 + k0 + 128, 0, 1);
            __builtin_prefetch(aRow1 + k0 + 128, 0, 1);
            __builtin_prefetch(bRow0 + k0 + 128, 0, 1);
            __builtin_prefetch(bRow2 + k0 + 128, 0, 1);
        }
#pragma unroll
        for (int si = 0; si < 2; ++si) {
            v16bf a = __builtin_bit_cast(v16bf, aU[si]);
#pragma unroll
            for (int j = 0; j < 4; ++j) {
                acc[si][j] = __builtin_amdgcn_wmma_f32_16x16x32_bf16(
                    false, a, false, __builtin_bit_cast(v16bf, bU[j]),
                    (short)0, acc[si][j], false, false);
            }
        }
    }

    // Epilogue: D element i -> (row M = i + 8*kg, col N = lane&15).
    // Reduce tanh(acc+hp)*wv over 16 columns; one partial per (nTile, row).
#pragma unroll
    for (int si = 0; si < 2; ++si) {
#pragma unroll
        for (int j = 0; j < 4; ++j) {
#pragma unroll
            for (int i = 0; i < 8; ++i) {
                float v = tanhf(acc[si][j][i] + hp[j]) * wv[j];
                v = half_sum16(v);
                if (ln == 0)
                    partial[(size_t)(nGroup * 4 + j) * S +
                            sGroup * 32 + si * 16 + 8 * kg + i] = v;
            }
        }
    }
}

// ---------------------------------------------------------------------------
// K6: scores[s] = sum_nt partial[nt, s] + b_va
// ---------------------------------------------------------------------------
__global__ void k_scores(const float* __restrict__ partial, const float* __restrict__ bva,
                         float* __restrict__ scores) {
    int s = blockIdx.x * blockDim.x + threadIdx.x;
    if (s >= S) return;
    float acc = 0.0f;
#pragma unroll 8
    for (int nt = 0; nt < NT; ++nt) acc += partial[(size_t)nt * S + s];
    scores[s] = acc + bva[0];
}

// ---------------------------------------------------------------------------
// K7: softmax over S (single block of 1024), write attn to d_out and ws
// ---------------------------------------------------------------------------
__global__ void k_softmax(const float* __restrict__ scores,
                          float* __restrict__ attn_out, float* __restrict__ attn_ws) {
    __shared__ float red[1024];
    const int tid = threadIdx.x;
    float m = -3.4e38f;
    for (int s = tid; s < S; s += 1024) m = fmaxf(m, scores[s]);
    red[tid] = m;
    __syncthreads();
    for (int w = 512; w > 0; w >>= 1) {
        if (tid < w) red[tid] = fmaxf(red[tid], red[tid + w]);
        __syncthreads();
    }
    const float mx = red[0];
    __syncthreads();
    float sum = 0.0f;
    for (int s = tid; s < S; s += 1024) sum += __expf(scores[s] - mx);
    red[tid] = sum;
    __syncthreads();
    for (int w = 512; w > 0; w >>= 1) {
        if (tid < w) red[tid] += red[tid + w];
        __syncthreads();
    }
    const float inv = 1.0f / red[0];
    for (int s = tid; s < S; s += 1024) {
        float a = __expf(scores[s] - mx) * inv;
        attn_out[s] = a;
        attn_ws[s]  = a;
    }
}

// ---------------------------------------------------------------------------
// K8: ct partials: block b covers rows [b*64, b*64+64); thread t covers 4 cols
// ---------------------------------------------------------------------------
__global__ void k_ct_part(const float* __restrict__ attn, const float* __restrict__ enc,
                          float* __restrict__ ctp) {
    const int b = blockIdx.x, t = threadIdx.x;   // 64 blocks x 256 threads
    float4 acc = {0.f, 0.f, 0.f, 0.f};
    for (int s = b * 64; s < b * 64 + 64; ++s) {
        float a = attn[s];
        float4 e = ((const float4*)(enc + (size_t)s * E))[t];
        acc.x += a * e.x; acc.y += a * e.y; acc.z += a * e.z; acc.w += a * e.w;
    }
    ((float4*)(ctp + (size_t)b * E))[t] = acc;
}

__global__ void k_ct_reduce(const float* __restrict__ ctp, float* __restrict__ ct) {
    int e = blockIdx.x * blockDim.x + threadIdx.x;
    if (e >= E) return;
    float acc = 0.0f;
#pragma unroll 8
    for (int b = 0; b < 64; ++b) acc += ctp[(size_t)b * E + e];
    ct[e] = acc;
}

// ---------------------------------------------------------------------------
// K9: ht_tilda = tanh(W_attn @ [ct; ht] + b_attn)   (one wave per row)
// ---------------------------------------------------------------------------
__global__ void k_httilda(const float* __restrict__ Wattn, const float* __restrict__ battn,
                          const float* __restrict__ ct, const float* __restrict__ ht,
                          float* __restrict__ httilda) {
    const int lane = threadIdx.x & 31;
    const int row  = (blockIdx.x * blockDim.x + threadIdx.x) >> 5;   // 0..255
    const float4* w4 = (const float4*)(Wattn + (size_t)row * (E + H));
    const float4* c4 = (const float4*)ct;
    const float4* h4 = (const float4*)ht;
    float acc = 0.0f;
#pragma unroll
    for (int j = 0; j < 8; ++j) {
        int idx = j * 32 + lane;
        acc += dot4(w4[idx], c4[idx]);
        acc += dot4(w4[256 + idx], h4[idx]);
    }
    acc = wave_sum32(acc);
    if (lane == 0) httilda[row] = tanhf(acc + battn[row]);
}

// ---------------------------------------------------------------------------
// K10: logits = W_out @ ht_tilda + b_out  (one wave per row, 32000 waves)
// ---------------------------------------------------------------------------
__global__ void k_logits(const float* __restrict__ Wout, const float* __restrict__ bout,
                         const float* __restrict__ httilda, float* __restrict__ logits) {
    const int lane = threadIdx.x & 31;
    const int row  = (blockIdx.x * blockDim.x + threadIdx.x) >> 5;   // 0..31999
    if (row >= V) return;
    const float4* w4 = (const float4*)(Wout + (size_t)row * H);
    const float4* h4 = (const float4*)httilda;
    float acc = 0.0f;
#pragma unroll
    for (int j = 0; j < 8; ++j) {
        int idx = j * 32 + lane;
        acc += dot4(w4[idx], h4[idx]);
    }
    acc = wave_sum32(acc);
    if (lane == 0) logits[row] = acc + bout[row];
}

// ---------------------------------------------------------------------------
// K11: log_softmax over V (single block of 1024)
// ---------------------------------------------------------------------------
__global__ void k_logsoftmax(const float* __restrict__ logits, float* __restrict__ out) {
    __shared__ float red[1024];
    const int tid = threadIdx.x;
    float m = -3.4e38f;
    for (int v = tid; v < V; v += 1024) m = fmaxf(m, logits[v]);
    red[tid] = m;
    __syncthreads();
    for (int w = 512; w > 0; w >>= 1) {
        if (tid < w) red[tid] = fmaxf(red[tid], red[tid + w]);
        __syncthreads();
    }
    const float mx = red[0];
    __syncthreads();
    float sum = 0.0f;
    for (int v = tid; v < V; v += 1024) sum += __expf(logits[v] - mx);
    red[tid] = sum;
    __syncthreads();
    for (int w = 512; w > 0; w >>= 1) {
        if (tid < w) red[tid] += red[tid + w];
        __syncthreads();
    }
    const float lse = mx + __logf(red[0]);
    for (int v = tid; v < V; v += 1024) out[v] = logits[v] - lse;
}

// ---------------------------------------------------------------------------
// Host launcher
// ---------------------------------------------------------------------------
extern "C" void kernel_launch(void* const* d_in, const int* in_sizes, int n_in,
                              void* d_out, int out_size, void* d_ws, size_t ws_size,
                              hipStream_t stream) {
    const int*   symbol = (const int*)d_in[0];
    const float* h0     = (const float*)d_in[1];
    const float* c0     = (const float*)d_in[2];
    const float* enc    = (const float*)d_in[3];
    const float* emb    = (const float*)d_in[4];
    const float* Wih    = (const float*)d_in[5];
    const float* bih    = (const float*)d_in[6];
    const float* Whh    = (const float*)d_in[7];
    const float* bhh    = (const float*)d_in[8];
    const float* Wscore = (const float*)d_in[9];
    const float* bscore = (const float*)d_in[10];
    const float* wva    = (const float*)d_in[11];
    const float* bva    = (const float*)d_in[12];
    const float* Wattn  = (const float*)d_in[13];
    const float* battn  = (const float*)d_in[14];
    const float* Wout   = (const float*)d_in[15];
    const float* bout   = (const float*)d_in[16];

    float* out = (float*)d_out;
    float* ws  = (float*)d_ws;
    unsigned short* bfb  = (unsigned short*)((char*)d_ws + WS_BF_BYTES);
    unsigned short* encb = bfb + WS_ENCB_OFF;
    unsigned short* wseb = bfb + WS_WSEB_OFF;

    // K1: LSTM gates (1024 waves)
    k_gates<<<128, 256, 0, stream>>>(symbol, emb, Wih, bih, Whh, bhh, h0, ws + WS_GATES);
    // K2: LSTM pointwise
    k_lstm_pw<<<4, 256, 0, stream>>>(ws + WS_GATES, c0, out + OUT_H, out + OUT_C, ws + WS_HT);
    // K3: hoisted ht-half of the score projection (256 waves)
    k_hpart<<<32, 256, 0, stream>>>(Wscore, bscore, ws + WS_HT, ws + WS_HPART);
    // K4: bf16 conversion of enc + W_score[:, H:]
    k_convert<<<2048, 256, 0, stream>>>(enc, Wscore, encb, wseb);
    // K5: WMMA attention GEMM, 32x64 macro-tile per wave (2048 waves)
    k_attn_gemm<<<256, 256, 0, stream>>>(encb, wseb, ws + WS_HPART, wva, ws + WS_PARTIAL);
    // K6: reduce partials -> scores
    k_scores<<<16, 256, 0, stream>>>(ws + WS_PARTIAL, bva, ws + WS_SCORES);
    // K7: softmax -> attn (d_out + ws)
    k_softmax<<<1, 1024, 0, stream>>>(ws + WS_SCORES, out + OUT_ATTN, ws + WS_ATTN);
    // K8/K8b: context vector ct = attn @ enc
    k_ct_part<<<64, 256, 0, stream>>>(ws + WS_ATTN, enc, ws + WS_CTPART);
    k_ct_reduce<<<4, 256, 0, stream>>>(ws + WS_CTPART, ws + WS_CT);
    // K9: attentional hidden (256 waves)
    k_httilda<<<32, 256, 0, stream>>>(Wattn, battn, ws + WS_CT, ws + WS_HT, ws + WS_HTILDA);
    // K10: output projection (32000 waves)
    k_logits<<<4000, 256, 0, stream>>>(Wout, bout, ws + WS_HTILDA, ws + WS_LOGITS);
    // K11: log_softmax -> d_out
    k_logsoftmax<<<1, 1024, 0, stream>>>(ws + WS_LOGITS, out + OUT_LOGP);
}
